// Attention_simple_12111807775229
// MI455X (gfx1250) — matrix operand order repaired and checked by reading, not compiled, not run on hardware
//
#include <hip/hip_runtime.h>

#define DDIM  1024
#define BATCH 32
#define SEQ   2048
#define ROWS  (BATCH * SEQ)

#define WPB 8            // waves per block (stage 1)
#define TM  16           // rows per wave tile
#define EB  8            // e-chunks (16 cols each) per A pass -> 24 WMMAs per A conversion
#define KB  256          // k-slice staged in LDS
#define LDSP 132         // dwords per W row in LDS: 132 % 64 == 4 -> conflict-minimal, 16B aligned
#define WPITCH 512       // dwords per W row in pre-split global planes (512 bf16 pairs)

typedef __attribute__((ext_vector_type(16))) __bf16        v16bf;
typedef __attribute__((ext_vector_type(8)))  float         v8f;
typedef __attribute__((ext_vector_type(4)))  unsigned int  v4u;

union BF16x16 {
    v16bf        bf;
    unsigned int u[8];
    v4u          q[2];
};

// pack two floats' bf16-truncations into one dword: {hi16(a), hi16(b)} (b -> low half)
__device__ inline unsigned pack_bf2(float a, float b) {
    return __builtin_amdgcn_perm(__float_as_uint(a), __float_as_uint(b), 0x07060302u);
}
__device__ inline float bf_trunc_f(float x) {
    return __uint_as_float(__float_as_uint(x) & 0xFFFF0000u);
}
__device__ inline float bf_res(float x) { return x - bf_trunc_f(x); }

// branch-free tanh: copysign(1 - 2/(e^{2|x|}+1), x); exp->inf saturates to 1
__device__ inline float fast_tanh(float x) {
    float ax = fabsf(x);
    float e2 = __expf(2.0f * ax);
    float y  = 1.0f - 2.0f / (e2 + 1.0f);
    return copysignf(y, x);
}

// ---------------------------------------------------------------------------
// Pre-pass: split W (fp32) once into packed bf16 hi/lo dword planes.
// ---------------------------------------------------------------------------
__global__ __launch_bounds__(256)
void split_w_kernel(const float* __restrict__ W,
                    unsigned int* __restrict__ whi,
                    unsigned int* __restrict__ wlo) {
    int i = blockIdx.x * 256 + threadIdx.x;        // pair index, 524288 total
    float2 w = ((const float2*)W)[i];
    whi[i] = pack_bf2(w.y, w.x);
    wlo[i] = pack_bf2(bf_res(w.y), bf_res(w.x));
}

// ---------------------------------------------------------------------------
// Stage 1: logits[b,s] = v . tanh(context[b,s,:] @ W^T)
// Per wave: 16-row tile. e blocked by EB*16=128 columns; pre-split W k-slices
// copied (no math) into padded LDS. fp32 ~= hi*hi + hi*lo + lo*hi via bf16 WMMA.
// ---------------------------------------------------------------------------
__global__ __launch_bounds__(256)
void attn_logits_kernel(const float* __restrict__ ctx,
                        const unsigned int* __restrict__ whi,
                        const unsigned int* __restrict__ wlo,
                        const float* __restrict__ v,
                        float* __restrict__ logits) {
    __shared__ unsigned int w_hi[16 * EB * LDSP];
    __shared__ unsigned int w_lo[16 * EB * LDSP];

    const int tid  = threadIdx.x;
    const int wave = tid >> 5;
    const int lane = tid & 31;
    const int half = lane >> 4;   // 0 / 1
    const int l15  = lane & 15;

    const int m0 = (blockIdx.x * WPB + wave) * TM;
    const float* arow = ctx + (size_t)(m0 + l15) * DDIM;

    float partial[8];
#pragma unroll
    for (int r = 0; r < 8; ++r) partial[r] = 0.f;

    for (int eb = 0; eb < DDIM; eb += 16 * EB) {
        v8f acc[EB];
#pragma unroll
        for (int e = 0; e < EB; ++e) acc[e] = (v8f){};

        for (int ks = 0; ks < DDIM; ks += KB) {
            __syncthreads();
            // Pure copy: pre-split W rows [eb, eb+128), dword cols [ks/2, ks/2+128) -> LDS
            {
                const int cs = ks >> 1;
                for (int idx = tid * 4; idx < 16 * EB * (KB / 2); idx += 256 * 4) {
                    int r = idx >> 7;          // / (KB/2)
                    int c = idx & (KB / 2 - 1);
                    size_t src = (size_t)(eb + r) * WPITCH + cs + c;
                    int    dst = r * LDSP + c;
                    *(v4u*)&w_hi[dst] = *(const v4u*)&whi[src];
                    *(v4u*)&w_lo[dst] = *(const v4u*)&wlo[src];
                }
            }
            __syncthreads();

            // Prefetch next A k-slice while we compute on this one
            __builtin_prefetch(arow + ks + KB, 0, 3);

            for (int kl = 0; kl < KB; kl += 32) {
                // ---- A fragment: lane l<16 row=l holds K kb..kb+7, kb+16..23 (kb=ks+kl);
                //      lane l>=16 same row, K shifted by 8 / 24.
                const float* ap = arow + ks + kl + half * 8;
                float4 s0 = *(const float4*)(ap);
                float4 s1 = *(const float4*)(ap + 4);
                float4 s2 = *(const float4*)(ap + 16);
                float4 s3 = *(const float4*)(ap + 20);

                BF16x16 a_hi, a_lo;
                a_hi.u[0] = pack_bf2(s0.y, s0.x);
                a_hi.u[1] = pack_bf2(s0.w, s0.z);
                a_hi.u[2] = pack_bf2(s1.y, s1.x);
                a_hi.u[3] = pack_bf2(s1.w, s1.z);
                a_hi.u[4] = pack_bf2(s2.y, s2.x);
                a_hi.u[5] = pack_bf2(s2.w, s2.z);
                a_hi.u[6] = pack_bf2(s3.y, s3.x);
                a_hi.u[7] = pack_bf2(s3.w, s3.z);
                a_lo.u[0] = pack_bf2(bf_res(s0.y), bf_res(s0.x));
                a_lo.u[1] = pack_bf2(bf_res(s0.w), bf_res(s0.z));
                a_lo.u[2] = pack_bf2(bf_res(s1.y), bf_res(s1.x));
                a_lo.u[3] = pack_bf2(bf_res(s1.w), bf_res(s1.z));
                a_lo.u[4] = pack_bf2(bf_res(s2.y), bf_res(s2.x));
                a_lo.u[5] = pack_bf2(bf_res(s2.w), bf_res(s2.z));
                a_lo.u[6] = pack_bf2(bf_res(s3.y), bf_res(s3.x));
                a_lo.u[7] = pack_bf2(bf_res(s3.w), bf_res(s3.z));

                const int bdw = (kl >> 1) + half * 4;   // dword offset in LDS row
#pragma unroll
                for (int e = 0; e < EB; ++e) {
                    // ---- B fragment: lane N=l15 -> W row (eb + e*16 + l15);
                    //      lanes 0-15: K kl..kl+7, kl+16..kl+23; lanes 16-31: K shifted by 8
                    const int rb = (e * 16 + l15) * LDSP + bdw;
                    BF16x16 b_hi, b_lo;
                    b_hi.q[0] = *(const v4u*)&w_hi[rb];
                    b_hi.q[1] = *(const v4u*)&w_hi[rb + 8];
                    b_lo.q[0] = *(const v4u*)&w_lo[rb];
                    b_lo.q[1] = *(const v4u*)&w_lo[rb + 8];

                    acc[e] = __builtin_amdgcn_wmma_f32_16x16x32_bf16(false, a_hi.bf, false, b_hi.bf,
                                                                     (short)0, acc[e], false, false);
                    acc[e] = __builtin_amdgcn_wmma_f32_16x16x32_bf16(false, a_hi.bf, false, b_lo.bf,
                                                                     (short)0, acc[e], false, false);
                    acc[e] = __builtin_amdgcn_wmma_f32_16x16x32_bf16(false, a_lo.bf, false, b_hi.bf,
                                                                     (short)0, acc[e], false, false);
                }
            }
        }

        // C layout: lane l<16 -> (M=r, N=l15); lane l>=16 -> (M=r+8, N=l15)
#pragma unroll
        for (int e = 0; e < EB; ++e) {
            float vn = v[eb + e * 16 + l15];
#pragma unroll
            for (int r = 0; r < 8; ++r) partial[r] += fast_tanh(acc[e][r]) * vn;
        }
    }

    // Reduce over N (16 lanes per half-wave)
#pragma unroll
    for (int r = 0; r < 8; ++r) {
        float s = partial[r];
        s += __shfl_xor(s, 1, 16);
        s += __shfl_xor(s, 2, 16);
        s += __shfl_xor(s, 4, 16);
        s += __shfl_xor(s, 8, 16);
        partial[r] = s;
    }
    if (l15 == 0) {
#pragma unroll
        for (int r = 0; r < 8; ++r)
            logits[m0 + half * 8 + r] = partial[r];
    }
}

// ---------------------------------------------------------------------------
// Stage 2: per-batch softmax over S, then weighted_context = attn^T @ context
// ---------------------------------------------------------------------------
__global__ __launch_bounds__(256)
void attn_softmax_weight_kernel(const float* __restrict__ ctx,
                                const float* __restrict__ logits,
                                const float* __restrict__ bias_p,
                                float* __restrict__ out_wctx,   // [B, D]
                                float* __restrict__ out_attn) { // [B, S]
    __shared__ float red[256];
    __shared__ float attn_s[SEQ];

    const int b   = blockIdx.x;
    const int tid = threadIdx.x;
    const float bias = bias_p[0];
    const float* lg = logits + (size_t)b * SEQ;

    float mx = -3.4e38f;
    for (int i = tid; i < SEQ; i += 256) mx = fmaxf(mx, lg[i] + bias);
    red[tid] = mx; __syncthreads();
    for (int off = 128; off > 0; off >>= 1) {
        if (tid < off) red[tid] = fmaxf(red[tid], red[tid + off]);
        __syncthreads();
    }
    mx = red[0]; __syncthreads();

    float sum = 0.f;
    for (int i = tid; i < SEQ; i += 256) {
        float e = __expf(lg[i] + bias - mx);
        attn_s[i] = e;
        sum += e;
    }
    red[tid] = sum; __syncthreads();
    for (int off = 128; off > 0; off >>= 1) {
        if (tid < off) red[tid] += red[tid + off];
        __syncthreads();
    }
    float inv = 1.f / red[0]; __syncthreads();

    for (int i = tid; i < SEQ; i += 256) {
        float a = attn_s[i] * inv;
        attn_s[i] = a;
        out_attn[(size_t)b * SEQ + i] = a;
    }
    __syncthreads();

    const float* cb = ctx + (size_t)b * SEQ * DDIM;
    const int d0 = tid * 4;
    float4 acc = {0.f, 0.f, 0.f, 0.f};
    for (int s = 0; s < SEQ; ++s) {
        float4 cv = *(const float4*)(cb + (size_t)s * DDIM + d0);
        float a = attn_s[s];
        acc.x += a * cv.x; acc.y += a * cv.y;
        acc.z += a * cv.z; acc.w += a * cv.w;
    }
    *(float4*)(out_wctx + (size_t)b * DDIM + d0) = acc;
}

extern "C" void kernel_launch(void* const* d_in, const int* in_sizes, int n_in,
                              void* d_out, int out_size, void* d_ws, size_t ws_size,
                              hipStream_t stream) {
    (void)in_sizes; (void)n_in; (void)out_size; (void)ws_size;
    const float* ctx = (const float*)d_in[0];   // [32,2048,1024]
    const float* W   = (const float*)d_in[1];   // [1024,1024]
    const float* v   = (const float*)d_in[2];   // [1024]
    const float* b   = (const float*)d_in[3];   // [1]

    float* out_wctx = (float*)d_out;                   // [32,1024]
    float* out_attn = (float*)d_out + BATCH * DDIM;    // [32,2048]

    // workspace: logits [65536 f32] | W_hi plane [1024*512 u32] | W_lo plane
    float*        logits = (float*)d_ws;
    unsigned int* whi    = (unsigned int*)((char*)d_ws + (size_t)ROWS * 4);
    unsigned int* wlo    = whi + (size_t)DDIM * WPITCH;

    dim3 blk(256);
    split_w_kernel<<<(DDIM * DDIM / 2) / 256, blk, 0, stream>>>(W, whi, wlo);
    attn_logits_kernel<<<ROWS / (TM * WPB), blk, 0, stream>>>(ctx, whi, wlo, v, logits);
    attn_softmax_weight_kernel<<<BATCH, blk, 0, stream>>>(ctx, logits, b, out_wctx, out_attn);
}
